// SharedBilinearDecoder_24584392802928
// MI455X (gfx1250) — matrix operand order repaired and checked by reading, not compiled, not run on hardware
//
#include <hip/hip_runtime.h>
#include <hip/hip_bf16.h>
#include <stdint.h>

typedef __attribute__((ext_vector_type(16))) __bf16        v16bf;
typedef __attribute__((ext_vector_type(8)))  float         v8f;
typedef __attribute__((ext_vector_type(8)))  unsigned int  v8u;

#define EMBED_DIM 256
#define NUM_W 8
#define NUM_R 5

__device__ __forceinline__ unsigned int pack2_bf16(float a, float b) {
  unsigned int ua = __builtin_bit_cast(unsigned int, a);
  unsigned int ub = __builtin_bit_cast(unsigned int, b);
  ua = (ua + 0x7FFFu + ((ua >> 16) & 1u)) >> 16;   // round-to-nearest-even
  ub = (ub + 0x7FFFu + ((ub >> 16) & 1u)) >> 16;
  return ua | (ub << 16);
}

// Stage 0: S_r = sum_w ws[w,r] * R_w, written as bf16 in exact WMMA B-fragment
// order: frag(r,n,kk) is 1024B = 32 lanes x 16 halves, lane(c,h) holds
// S_r[K = 32*kk + 16*h + t][N = 16*n + c] for t = 0..15.
__global__ __launch_bounds__(256) void build_sfrag_kernel(
    const float* __restrict__ rel, const float* __restrict__ ws,
    unsigned short* __restrict__ sfrag)
{
  unsigned int gid  = blockIdx.x * 256u + threadIdx.x;   // < 5*65536
  unsigned int t    = gid & 15u;
  unsigned int lane = (gid >> 4) & 31u;
  unsigned int kk   = (gid >> 9) & 7u;
  unsigned int n    = (gid >> 12) & 15u;
  unsigned int r    = gid >> 16;
  unsigned int c = lane & 15u, h = lane >> 4;
  unsigned int K = kk * 32u + h * 16u + t;
  unsigned int N = n * 16u + c;
  float acc = 0.f;
#pragma unroll
  for (int w = 0; w < NUM_W; ++w)
    acc += ws[w * NUM_R + r] * rel[(size_t)w * (EMBED_DIM * EMBED_DIM) + K * EMBED_DIM + N];
  unsigned int u = __builtin_bit_cast(unsigned int, acc);
  sfrag[gid] = (unsigned short)((u + 0x7FFFu + ((u >> 16) & 1u)) >> 16);
}

// Async-copy one 128KB S_r fragment block (global, fragment-ordered) into LDS.
// 8192 16B chunks; thread t copies chunks {j*256 + t}: at fixed j the wave
// reads 32 consecutive 16B chunks -> fully coalesced. ASYNCcnt-tracked,
// writes LDS directly (no VGPR round trip).
__device__ __forceinline__ void async_copy_sr(const unsigned int* __restrict__ gsrc,
                                              unsigned lds_base, int tid) {
#pragma unroll
  for (int j = 0; j < 32; ++j) {
    unsigned off = (unsigned)(j * 4096) + (unsigned)tid * 16u;
    const void* g = (const char*)gsrc + off;
    unsigned l = lds_base + off;
    asm volatile("global_load_async_to_lds_b128 %0, %1, off"
                 :: "v"(l), "v"(g) : "memory");
  }
}

// Main: each wave owns 32 batch rows (two 16-row M-tiles sharing every
// B fragment); block (8 waves) owns 256 rows. S_r double-buffered in LDS.
__global__ __launch_bounds__(256) void bilinear_wmma_kernel(
    const float* __restrict__ e1, const float* __restrict__ e2,
    const unsigned int* __restrict__ sfragU, float* __restrict__ out)
{
  __shared__ v8u sbuf[2][4096];   // 2 x 128KB

  const int tid  = threadIdx.x;
  const int lane = tid & 31;
  const int wave = tid >> 5;
  const int h = lane >> 4;          // lane half
  const int c = lane & 15;          // lane within half
  const size_t row0 = (size_t)blockIdx.x * 256u + (size_t)wave * 32u;

  // Kick off the r=0 staging before the (long) A-fragment gather.
  const unsigned lds0 = (unsigned)(uintptr_t)(void*)&sbuf[0][0];
  const unsigned lds1 = (unsigned)(uintptr_t)(void*)&sbuf[1][0];
  async_copy_sr(sfragU, lds0, tid);

  // ---- Preload A fragments: two 16x256 e1 tiles per wave (bf16). ----
  // 16-bit A 16x32 layout (contiguous, matches B frags): lanes 0-15 (h=0): K = k0+0..15;
  // lanes 16-31 (h=1): K = k0+16..31.
  v16bf afrag[2][8];
#pragma unroll
  for (int m = 0; m < 2; ++m) {
    const size_t arow = row0 + (size_t)(16 * m) + (size_t)c;
#pragma unroll
    for (int kk = 0; kk < 8; ++kk) {
      const float4* p = (const float4*)(e1 + arow * EMBED_DIM + kk * 32 + 16 * h);
      float4 g0 = p[0];
      float4 g1 = p[1];
      float4 g2 = p[2];   // +8 floats
      float4 g3 = p[3];
      v8u au;
      au[0] = pack2_bf16(g0.x, g0.y);
      au[1] = pack2_bf16(g0.z, g0.w);
      au[2] = pack2_bf16(g1.x, g1.y);
      au[3] = pack2_bf16(g1.z, g1.w);
      au[4] = pack2_bf16(g2.x, g2.y);
      au[5] = pack2_bf16(g2.z, g2.w);
      au[6] = pack2_bf16(g3.x, g3.y);
      au[7] = pack2_bf16(g3.z, g3.w);
      afrag[m][kk] = __builtin_bit_cast(v16bf, au);
    }
  }

  // Per-lane e2/out bases for the two M-tiles (row = base + v*... , col c).
  const float* e2b0 = e2 + (row0 + (size_t)(8 * h)) * EMBED_DIM + c;
  const float* e2b1 = e2 + (row0 + (size_t)(16 + 8 * h)) * EMBED_DIM + c;

  for (int r = 0; r < NUM_R; ++r) {
    // Wait for this r's staging (only outstanding asyncs), then sync waves.
    asm volatile("s_wait_asynccnt 0x0" ::: "memory");
    __syncthreads();
    // Overlap next r's staging with this r's compute. Target buffer was last
    // read in the previous iteration; the trailing barrier of r-1 protects it.
    if (r + 1 < NUM_R)
      async_copy_sr(sfragU + (size_t)(r + 1) * 32768u,
                    ((r + 1) & 1) ? lds1 : lds0, tid);

    const v8u* bbuf = &sbuf[r & 1][0];
    float part0[8], part1[8];
#pragma unroll
    for (int v = 0; v < 8; ++v) { part0[v] = 0.f; part1[v] = 0.f; }

    for (int n = 0; n < 16; ++n) {
      v8f acc0 = {0.f, 0.f, 0.f, 0.f, 0.f, 0.f, 0.f, 0.f};
      v8f acc1 = {0.f, 0.f, 0.f, 0.f, 0.f, 0.f, 0.f, 0.f};
#pragma unroll
      for (int kk = 0; kk < 8; ++kk) {
        v8u b = bbuf[(n * 8 + kk) * 32 + lane];     // conflict-free ds_load_b128
        v16bf bf = __builtin_bit_cast(v16bf, b);
        acc0 = __builtin_amdgcn_wmma_f32_16x16x32_bf16(
            false, afrag[0][kk], false, bf, (short)0, acc0, false, false);
        acc1 = __builtin_amdgcn_wmma_f32_16x16x32_bf16(
            false, afrag[1][kk], false, bf, (short)0, acc1, false, false);
      }
      // D layout: VGPR v, half h -> row v+8h, col c (+16n). Dot with e2.
#pragma unroll
      for (int v = 0; v < 8; ++v) {
        part0[v] += acc0[v] * e2b0[(size_t)v * EMBED_DIM + n * 16];
        part1[v] += acc1[v] * e2b1[(size_t)v * EMBED_DIM + n * 16];
      }
    }

    // Reduce across the 16 lanes of each half (xor masks < 16 stay in-half).
#pragma unroll
    for (int v = 0; v < 8; ++v) {
      float p0 = part0[v], p1 = part1[v];
      p0 += __shfl_xor(p0, 1);  p1 += __shfl_xor(p1, 1);
      p0 += __shfl_xor(p0, 2);  p1 += __shfl_xor(p1, 2);
      p0 += __shfl_xor(p0, 4);  p1 += __shfl_xor(p1, 4);
      p0 += __shfl_xor(p0, 8);  p1 += __shfl_xor(p1, 8);
      part0[v] = p0;  part1[v] = p1;
    }
    if (c == 0) {
#pragma unroll
      for (int v = 0; v < 8; ++v) {
        out[(row0 + (size_t)(v + 8 * h)) * NUM_R + r]      = part0[v];
        out[(row0 + (size_t)(16 + v + 8 * h)) * NUM_R + r] = part1[v];
      }
    }
    __syncthreads();   // all waves done reading sbuf[r&1] before it's refilled
  }
}

extern "C" void kernel_launch(void* const* d_in, const int* in_sizes, int n_in,
                              void* d_out, int out_size, void* d_ws, size_t ws_size,
                              hipStream_t stream) {
  const float* e1  = (const float*)d_in[0];   // [B, 256]
  const float* e2  = (const float*)d_in[1];   // [B, 256]
  const float* rel = (const float*)d_in[2];   // [8, 256*256]
  const float* ws  = (const float*)d_in[3];   // [8, 5]
  float* out = (float*)d_out;                 // [B, 5]
  unsigned short* sfrag = (unsigned short*)d_ws;   // 5*256*256 bf16 = 640 KB

  const int batch = in_sizes[0] / EMBED_DIM;  // 131072

  build_sfrag_kernel<<<(NUM_R * 65536) / 256, 256, 0, stream>>>(rel, ws, sfrag);

  // Main: 256 batch rows per block (8 waves x 32 rows).
  bilinear_wmma_kernel<<<batch / 256, 256, 0, stream>>>(
      e1, e2, (const unsigned int*)sfrag, out);
}
